// GPTModel_50130858279059
// MI455X (gfx1250) — matrix operand order repaired and checked by reading, not compiled, not run on hardware
//
#include <hip/hip_runtime.h>
#include <hip/hip_bf16.h>
#include <math.h>

// ---- model constants (GPT-2 small, B=2, T=1024) ----
#define VOCAB  50257
#define NLAYER 12
#define NHEAD  12
#define DMODEL 768
#define SEQ    1024
#define FFDIM  3072
#define HEADD  64
#define BATCH  2
#define MROWS  (BATCH * SEQ)   // 2048
#define BPAD   40              // padded n-row stride (u16) for B tile in LDS, 16B aligned

typedef unsigned short u16;
typedef __bf16 v16bf __attribute__((ext_vector_type(16)));
typedef float  v8f   __attribute__((ext_vector_type(8)));

union BFrag { v16bf v; u16 u[16]; uint4 q[2]; };

__device__ inline u16 f2bf(float f) {
    unsigned u = __float_as_uint(f);
    unsigned r = u + 0x7FFFu + ((u >> 16) & 1u);   // round-to-nearest-even
    return (u16)(r >> 16);
}

__device__ inline v8f zero8() {
    v8f z;
#pragma unroll
    for (int i = 0; i < 8; ++i) z[i] = 0.f;
    return z;
}

// ---------------- fp32 -> bf16 bulk convert ----------------
__global__ void cvt_kernel(const float* __restrict__ in, u16* __restrict__ out, size_t n) {
    size_t i = (size_t)blockIdx.x * blockDim.x + threadIdx.x;
    size_t stride = (size_t)gridDim.x * blockDim.x;
    for (; i < n; i += stride) out[i] = f2bf(in[i]);
}

// ---------------- embedding gather ----------------
__global__ void embed_kernel(const int* __restrict__ ids, const float* __restrict__ tok,
                             const float* __restrict__ pos, float* __restrict__ x) {
    size_t n = (size_t)MROWS * DMODEL;
    size_t i = (size_t)blockIdx.x * blockDim.x + threadIdx.x;
    size_t stride = (size_t)gridDim.x * blockDim.x;
    for (; i < n; i += stride) {
        int m = (int)(i / DMODEL), d = (int)(i % DMODEL);
        int t = m % SEQ;
        x[i] = tok[(size_t)ids[m] * DMODEL + d] + pos[(size_t)t * DMODEL + d];
    }
}

// ---------------- LayerNorm: fp32 in -> bf16 out ----------------
__global__ __launch_bounds__(256)
void ln_kernel(const float* __restrict__ x, const float* __restrict__ g,
               const float* __restrict__ b, u16* __restrict__ out) {
    __shared__ float s1[256], s2[256];
    int row = blockIdx.x, tid = threadIdx.x;
    const float* xr = x + (size_t)row * DMODEL;
    float s = 0.f, ss = 0.f;
    for (int i = tid; i < DMODEL; i += 256) { float v = xr[i]; s += v; ss += v * v; }
    s1[tid] = s; s2[tid] = ss;
    __syncthreads();
    for (int o = 128; o > 0; o >>= 1) {
        if (tid < o) { s1[tid] += s1[tid + o]; s2[tid] += s2[tid + o]; }
        __syncthreads();
    }
    float mean = s1[0] / DMODEL;
    float var  = s2[0] / DMODEL - mean * mean;
    float rstd = rsqrtf(var + 1e-5f);
    u16* orow = out + (size_t)row * DMODEL;
    for (int i = tid; i < DMODEL; i += 256)
        orow[i] = f2bf((xr[i] - mean) * rstd * g[i] + b[i]);
}

// ---- split fused QKV (fp32) into per-head bf16 Q[t][d], K[t][d], and V TRANSPOSED [d][t] ----
__global__ void split_kernel(const float* __restrict__ qkv, u16* __restrict__ q,
                             u16* __restrict__ k, u16* __restrict__ v) {
    size_t n = (size_t)MROWS * DMODEL;
    size_t i = (size_t)blockIdx.x * blockDim.x + threadIdx.x;
    size_t stride = (size_t)gridDim.x * blockDim.x;
    for (; i < n; i += stride) {
        int m = (int)(i / DMODEL), c = (int)(i % DMODEL);
        int bb = m / SEQ, t = m % SEQ, hh = c / HEADD, dd = c % HEADD;
        int bh = bb * NHEAD + hh;
        size_t o  = ((size_t)bh * SEQ + t) * HEADD + dd;       // [bh][t][d]
        size_t ov = ((size_t)bh * HEADD + dd) * SEQ + t;       // [bh][d][t] (transposed)
        const float* row = qkv + (size_t)m * (3 * DMODEL);
        q[o]  = f2bf(row[c]);
        k[o]  = f2bf(row[DMODEL + c]);
        v[ov] = f2bf(row[2 * DMODEL + c]);
    }
}

// ---------------- WMMA GEMM: C = A(bf16 MxK) * B(bf16 KxN, or NxK if btrans) ----------------
// 128 threads = 4 waves; 128x64 block tile, K-step 32; wave w -> rows [32w,32w+32), all 64 cols.
// Per k-step per wave: 2 A frags + 4 B frags -> 8 WMMAs (A reused x4, B reused x2).
// LDS: A tile K-major (A frag = 2x b128); B tile n-major padded (B frag = 2x b128).
__global__ __launch_bounds__(128)
void wmma_gemm_kernel(const u16* __restrict__ A, const u16* __restrict__ B,
                      const float* __restrict__ bias, const float* __restrict__ res,
                      void* __restrict__ outp, int out_bf16, int do_gelu, int btrans,
                      int M, int N, int K) {
    __shared__ __align__(16) u16 As[128 * 32];
    __shared__ __align__(16) u16 Bs[64 * BPAD];   // Bs[n*BPAD + k], n<64, k<32
    const int tid = threadIdx.x;
    const int w = tid >> 5, lane = tid & 31, hf = lane >> 4, ln = lane & 15;
    const int bm = blockIdx.y * 128, bn = blockIdx.x * 64;

    v8f acc[2][4];
    for (int mi = 0; mi < 2; ++mi)
        for (int f = 0; f < 4; ++f) acc[mi][f] = zero8();

    const int arow = tid >> 1, acol = (tid & 1) * 16;     // A staging: 2x (64 rows x 32 cols)
    const int brow = tid & 31, bcol = (tid >> 5) * 16;    // B staging (KxN source)
    const int tn = tid >> 1,  tkc  = (tid & 1) * 16;      // B staging (NxK transposed source)

    for (int k0 = 0; k0 < K; k0 += 32) {
        // ---- A tile (128x32) -> LDS, two row-chunks per thread ----
#pragma unroll
        for (int c = 0; c < 2; ++c) {
            int r = arow + c * 64;
            int gm = bm + r;
            uint4 v0 = {0,0,0,0}, v1 = {0,0,0,0};
            if (gm < M) {
                const uint4* p = reinterpret_cast<const uint4*>(A + (size_t)gm * K + k0 + acol);
                v0 = p[0]; v1 = p[1];
            }
            uint4* s = reinterpret_cast<uint4*>(&As[r * 32 + acol]);
            s[0] = v0; s[1] = v1;
        }
        if (!btrans) { // ---- B tile (KxN): vector global load, scatter to n-major LDS ----
            int gk = k0 + brow;                 // k row, 0..31
            int gn = bn + bcol;                 // 16-col chunk
            BFrag tmp;
            if (gn + 16 <= N) {
                const uint4* p = reinterpret_cast<const uint4*>(B + (size_t)gk * N + gn);
                tmp.q[0] = p[0]; tmp.q[1] = p[1];
            } else {
#pragma unroll
                for (int j = 0; j < 16; ++j)
                    tmp.u[j] = (gn + j < N) ? B[(size_t)gk * N + gn + j] : (u16)0;
            }
#pragma unroll
            for (int j = 0; j < 16; ++j)
                Bs[(bcol + j) * BPAD + brow] = tmp.u[j];   // consecutive lanes -> consecutive u16
        } else {       // ---- B tile from NxK (tok_emb LM head): fully vectorized ----
            int gn = bn + tn;
            BFrag tmp;
            if (gn < N) {
                const uint4* p = reinterpret_cast<const uint4*>(B + (size_t)gn * K + k0 + tkc);
                tmp.q[0] = p[0]; tmp.q[1] = p[1];
            } else {
                tmp.q[0] = uint4{0,0,0,0}; tmp.q[1] = uint4{0,0,0,0};
            }
            uint4* s = reinterpret_cast<uint4*>(&Bs[tn * BPAD + tkc]);
            s[0] = tmp.q[0]; s[1] = tmp.q[1];
        }
        __syncthreads();

        if (k0 + 32 < K) { // prefetch next tiles (global_prefetch_b8)
            int gm = bm + arow;
            if (gm < M) __builtin_prefetch(A + (size_t)gm * K + k0 + 32 + acol, 0, 0);
            if (!btrans) __builtin_prefetch(B + (size_t)(k0 + 32 + brow) * N + bn + bcol, 0, 0);
            else if (bn + tn < N) __builtin_prefetch(B + (size_t)(bn + tn) * K + k0 + 32 + tkc, 0, 0);
        }

        // A fragments (ISA 16-bit A layout: lane=m, K = {8h..8h+7} U {16+8h..16+8h+7})
        BFrag af[2];
#pragma unroll
        for (int mi = 0; mi < 2; ++mi) {
            const uint4* ar = reinterpret_cast<const uint4*>(&As[(w * 32 + mi * 16 + ln) * 32]);
            af[mi].q[0] = ar[hf];        // elements 8*hf .. 8*hf+7
            af[mi].q[1] = ar[2 + hf];    // elements 16+8*hf .. 16+8*hf+7
        }
#pragma unroll
        for (int f = 0; f < 4; ++f) {
            BFrag bf; // B layout: lane=n, K = {8h..8h+7} U {16+8h..16+8h+7} (matches A)
            const uint4* br = reinterpret_cast<const uint4*>(&Bs[(f * 16 + ln) * BPAD]);
            bf.q[0] = br[hf]; bf.q[1] = br[2 + hf];
#pragma unroll
            for (int mi = 0; mi < 2; ++mi)
                acc[mi][f] = __builtin_amdgcn_wmma_f32_16x16x32_bf16(false, af[mi].v, false, bf.v,
                                                                     (short)0, acc[mi][f], false, false);
        }
        __syncthreads();
    }

    // epilogue: C/D layout lane=n, M = r + 8*half
#pragma unroll
    for (int mi = 0; mi < 2; ++mi) {
#pragma unroll
        for (int f = 0; f < 4; ++f) {
#pragma unroll
            for (int r = 0; r < 8; ++r) {
                int gm = bm + w * 32 + mi * 16 + r + 8 * hf;
                int gn = bn + f * 16 + ln;
                if (gm < M && gn < N) {
                    float vv = acc[mi][f][r];
                    if (bias) vv += bias[gn];
                    if (do_gelu) vv = 0.5f * vv * (1.f + erff(vv * 0.70710678118654752f));
                    if (res) vv += res[(size_t)gm * N + gn];
                    if (out_bf16) ((u16*)outp)[(size_t)gm * N + gn] = f2bf(vv);
                    else          ((float*)outp)[(size_t)gm * N + gn] = vv;
                }
            }
        }
    }
}

// ---------------- flash attention: one wave per (b,h, 16 query rows) ----------------
// Q,K are [bh][t][d]; V is [bh][d][t] (pre-transposed) so every fragment is 2x b128.
__global__ __launch_bounds__(32)
void attn_kernel(const u16* __restrict__ qb, const u16* __restrict__ kb,
                 const u16* __restrict__ vb, u16* __restrict__ attn) {
    __shared__ __align__(16) u16 lp[16 * 32];
    const int lane = threadIdx.x, hf = lane >> 4, ln = lane & 15;
    const int bh = blockIdx.y, b = bh / NHEAD, h = bh % NHEAD;
    const int row0 = blockIdx.x * 16;
    const u16* q = qb + (size_t)bh * SEQ * HEADD;
    const u16* k = kb + (size_t)bh * SEQ * HEADD;
    const u16* v = vb + (size_t)bh * HEADD * SEQ;

    BFrag aq[2]; // Q 16x64 as two K=32 A fragments (vector loads, 16B aligned by construction)
#pragma unroll
    for (int kd = 0; kd < 2; ++kd) {
        const u16* qp = q + (size_t)(row0 + ln) * HEADD + kd * 32;
        aq[kd].q[0] = *reinterpret_cast<const uint4*>(qp + 8 * hf);
        aq[kd].q[1] = *reinterpret_cast<const uint4*>(qp + 16 + 8 * hf);
    }

    float Mr[8], Sr[8];
    v8f O[4];
#pragma unroll
    for (int r = 0; r < 8; ++r) { Mr[r] = -1e30f; Sr[r] = 0.f; }
    for (int f = 0; f < 4; ++f) O[f] = zero8();

    const int nkt = (row0 + 16 + 31) / 32; // causal: keys <= row0+15
    for (int kt = 0; kt < nkt; ++kt) {
        int key0 = kt * 32;
        // ---- S = Q * K^T (16 x 32 keys), two N tiles, K-dim = HEADD = 2x32 ----
        v8f sc[2]; sc[0] = zero8(); sc[1] = zero8();
#pragma unroll
        for (int t = 0; t < 2; ++t) {
            int key = key0 + t * 16 + ln;
#pragma unroll
            for (int kd = 0; kd < 2; ++kd) {
                BFrag bk;
                const u16* kp = k + (size_t)key * HEADD + kd * 32 + 8 * hf;
                bk.q[0] = *reinterpret_cast<const uint4*>(kp);
                bk.q[1] = *reinterpret_cast<const uint4*>(kp + 16);
                sc[t] = __builtin_amdgcn_wmma_f32_16x16x32_bf16(false, aq[kd].v, false, bk.v,
                                                                (short)0, sc[t], false, false);
            }
        }
        // ---- scale + causal mask, online softmax (rows live in lane half-groups) ----
        float p[2][8], rm[8];
#pragma unroll
        for (int r = 0; r < 8; ++r) rm[r] = -1e30f;
#pragma unroll
        for (int t = 0; t < 2; ++t)
#pragma unroll
            for (int r = 0; r < 8; ++r) {
                int rg = row0 + r + 8 * hf;
                int cg = key0 + t * 16 + ln;
                float vs = sc[t][r] * 0.125f;          // 1/sqrt(64)
                if (cg > rg) vs = -1e30f;
                p[t][r] = vs;
                rm[r] = fmaxf(rm[r], vs);
            }
#pragma unroll
        for (int r = 0; r < 8; ++r) {
            for (int o = 1; o < 16; o <<= 1) rm[r] = fmaxf(rm[r], __shfl_xor(rm[r], o, 32));
            float nm = fmaxf(Mr[r], rm[r]);
            float al = __expf(Mr[r] - nm);
            Mr[r] = nm;
            p[0][r] = __expf(p[0][r] - nm);
            p[1][r] = __expf(p[1][r] - nm);
            float ps = p[0][r] + p[1][r];
            for (int o = 1; o < 16; o <<= 1) ps += __shfl_xor(ps, o, 32);
            Sr[r] = Sr[r] * al + ps;
#pragma unroll
            for (int f = 0; f < 4; ++f) O[f][r] *= al;
        }
        // ---- C-layout -> A-layout via LDS (single wave: LDS ops are in-order) ----
#pragma unroll
        for (int t = 0; t < 2; ++t)
#pragma unroll
            for (int r = 0; r < 8; ++r)
                lp[(r + 8 * hf) * 32 + t * 16 + ln] = f2bf(p[t][r]);
        BFrag ap;
        {
            const uint4* pr = reinterpret_cast<const uint4*>(&lp[ln * 32]);
            ap.q[0] = pr[hf];
            ap.q[1] = pr[2 + hf];
        }
        // ---- O += P * V (K = 32 keys, N = 64 head dims); V transposed -> contiguous ----
#pragma unroll
        for (int f = 0; f < 4; ++f) {
            BFrag bv;
            const u16* vp = v + (size_t)(f * 16 + ln) * SEQ + key0 + 8 * hf;
            bv.q[0] = *reinterpret_cast<const uint4*>(vp);
            bv.q[1] = *reinterpret_cast<const uint4*>(vp + 16);
            O[f] = __builtin_amdgcn_wmma_f32_16x16x32_bf16(false, ap.v, false, bv.v,
                                                           (short)0, O[f], false, false);
        }
    }
    // ---- normalize and scatter back to (B,T,D) bf16 ----
#pragma unroll
    for (int f = 0; f < 4; ++f)
#pragma unroll
        for (int r = 0; r < 8; ++r) {
            int rg = row0 + r + 8 * hf;
            float ov = O[f][r] / Sr[r];
            attn[((size_t)(b * SEQ + rg)) * DMODEL + h * HEADD + f * 16 + ln] = f2bf(ov);
        }
}

// =====================================================================
extern "C" void kernel_launch(void* const* d_in, const int* in_sizes, int n_in,
                              void* d_out, int out_size, void* d_ws, size_t ws_size,
                              hipStream_t stream) {
    (void)in_sizes; (void)n_in; (void)out_size; (void)ws_size;
    const int*   ids  = (const int*)  d_in[0];
    const float* tok  = (const float*)d_in[1];
    const float* pos  = (const float*)d_in[2];
    const float* ln1w = (const float*)d_in[3];
    const float* ln1b = (const float*)d_in[4];
    const float* qkvw = (const float*)d_in[5];
    const float* qkvb = (const float*)d_in[6];
    const float* outw = (const float*)d_in[7];
    const float* outb = (const float*)d_in[8];
    const float* ln2w = (const float*)d_in[9];
    const float* ln2b = (const float*)d_in[10];
    const float* f1w  = (const float*)d_in[11];
    const float* f1b  = (const float*)d_in[12];
    const float* f2w  = (const float*)d_in[13];
    const float* f2b  = (const float*)d_in[14];
    const float* lnfw = (const float*)d_in[15];
    const float* lnfb = (const float*)d_in[16];

    char* ws = (char*)d_ws;
    size_t off = 0;
    auto take = [&](size_t bytes) {
        size_t o = (off + 255) & ~(size_t)255;
        off = o + bytes;
        return o;
    };

    u16*   tokw16 = (u16*)  (ws + take((size_t)VOCAB * DMODEL * 2));
    u16*   qkvw16 = (u16*)  (ws + take((size_t)NLAYER * DMODEL * 3 * DMODEL * 2));
    u16*   outw16 = (u16*)  (ws + take((size_t)NLAYER * DMODEL * DMODEL * 2));
    u16*   f1w16  = (u16*)  (ws + take((size_t)NLAYER * DMODEL * FFDIM * 2));
    u16*   f2w16  = (u16*)  (ws + take((size_t)NLAYER * FFDIM * DMODEL * 2));
    float* xbuf   = (float*)(ws + take((size_t)MROWS * DMODEL * 4));
    u16*   hbuf   = (u16*)  (ws + take((size_t)MROWS * DMODEL * 2));
    float* qkvbuf = (float*)(ws + take((size_t)MROWS * 3 * DMODEL * 4));
    u16*   qh     = (u16*)  (ws + take((size_t)MROWS * DMODEL * 2));
    u16*   kh     = (u16*)  (ws + take((size_t)MROWS * DMODEL * 2));
    u16*   vh     = (u16*)  (ws + take((size_t)MROWS * DMODEL * 2));
    u16*   attnb  = (u16*)  (ws + take((size_t)MROWS * DMODEL * 2));
    u16*   ffbuf  = (u16*)  (ws + take((size_t)MROWS * FFDIM * 2));

    // weight conversions (fp32 -> bf16), deterministic every call
    cvt_kernel<<<2048, 256, 0, stream>>>(tok,  tokw16, (size_t)VOCAB * DMODEL);
    cvt_kernel<<<2048, 256, 0, stream>>>(qkvw, qkvw16, (size_t)NLAYER * DMODEL * 3 * DMODEL);
    cvt_kernel<<<2048, 256, 0, stream>>>(outw, outw16, (size_t)NLAYER * DMODEL * DMODEL);
    cvt_kernel<<<2048, 256, 0, stream>>>(f1w,  f1w16,  (size_t)NLAYER * DMODEL * FFDIM);
    cvt_kernel<<<2048, 256, 0, stream>>>(f2w,  f2w16,  (size_t)NLAYER * FFDIM * DMODEL);

    embed_kernel<<<2048, 256, 0, stream>>>(ids, tok, pos, xbuf);

    const dim3 gemmBlk(128);
    for (int l = 0; l < NLAYER; ++l) {
        ln_kernel<<<MROWS, 256, 0, stream>>>(xbuf, ln1w + l * DMODEL, ln1b + l * DMODEL, hbuf);
        wmma_gemm_kernel<<<dim3(3 * DMODEL / 64, MROWS / 128), gemmBlk, 0, stream>>>(
            hbuf, qkvw16 + (size_t)l * DMODEL * 3 * DMODEL, qkvb + l * 3 * DMODEL,
            nullptr, qkvbuf, 0, 0, 0, MROWS, 3 * DMODEL, DMODEL);
        split_kernel<<<2048, 256, 0, stream>>>(qkvbuf, qh, kh, vh);
        attn_kernel<<<dim3(SEQ / 16, BATCH * NHEAD), 32, 0, stream>>>(qh, kh, vh, attnb);
        wmma_gemm_kernel<<<dim3(DMODEL / 64, MROWS / 128), gemmBlk, 0, stream>>>(
            attnb, outw16 + (size_t)l * DMODEL * DMODEL, outb + l * DMODEL,
            xbuf, xbuf, 0, 0, 0, MROWS, DMODEL, DMODEL);
        ln_kernel<<<MROWS, 256, 0, stream>>>(xbuf, ln2w + l * DMODEL, ln2b + l * DMODEL, hbuf);
        wmma_gemm_kernel<<<dim3(FFDIM / 64, MROWS / 128), gemmBlk, 0, stream>>>(
            hbuf, f1w16 + (size_t)l * DMODEL * FFDIM, f1b + l * FFDIM,
            nullptr, ffbuf, 1, 1, 0, MROWS, FFDIM, DMODEL);
        wmma_gemm_kernel<<<dim3(DMODEL / 64, MROWS / 128), gemmBlk, 0, stream>>>(
            ffbuf, f2w16 + (size_t)l * FFDIM * DMODEL, f2b + l * DMODEL,
            xbuf, xbuf, 0, 0, 0, MROWS, DMODEL, FFDIM);
    }

    ln_kernel<<<MROWS, 256, 0, stream>>>(xbuf, lnfw, lnfb, hbuf);
    // tied LM head: logits = h @ tok_emb^T (B supplied as NxK, btrans=1)
    wmma_gemm_kernel<<<dim3((VOCAB + 63) / 64, MROWS / 128), gemmBlk, 0, stream>>>(
        hbuf, tokw16, nullptr, nullptr, d_out, 0, 0, 1, MROWS, VOCAB, DMODEL);
}
